// SelfAttention_1176821039582
// MI455X (gfx1250) — matrix operand order repaired and checked by reading, not compiled, not run on hardware
//
#include <hip/hip_runtime.h>

// Problem constants (match reference)
#define B_SZ   64
#define C_DIM  512
#define N_DIM  512
#define OUT_DIM 512
#define GROUPS 32
#define EPS_GN 1e-5f

typedef __bf16 bf16_t;
typedef __attribute__((ext_vector_type(16))) __bf16 v16bf;
typedef __attribute__((ext_vector_type(8)))  float  v8f;
typedef __attribute__((ext_vector_type(4)))  unsigned int u32x4;
typedef __attribute__((ext_vector_type(2)))  float  f32x2;

union FragU { v16bf v; u32x4 q[2]; };

__device__ __forceinline__ v8f wmma_bf16(v16bf a, v16bf b, v8f c) {
  // D = A(16x32) * B(32x16) + C, f32 accumulate
  return __builtin_amdgcn_wmma_f32_16x16x32_bf16(
      false, a, false, b, (short)0, c, false, false);
}

// A fragment: 16x32 tile of row-major (rows x K) bf16 matrix, leading dim ld.
// lane L: row = L&15, kb = (L>>4)*8; halves 0..7 -> K=kb..kb+7, 8..15 -> K=kb+16..kb+23
__device__ __forceinline__ v16bf load_frag_a(const bf16_t* base, int ld) {
  int lane = threadIdx.x & 31;
  int row  = lane & 15;
  int kb   = (lane >> 4) << 3;
  const bf16_t* p = base + (size_t)row * ld + kb;
  FragU u;
  u.q[0] = *reinterpret_cast<const u32x4*>(p);
  u.q[1] = *reinterpret_cast<const u32x4*>(p + 16);
  return u.v;
}

// B fragment: 32x16 (K x n) tile sourced from row-major S(n, K), leading dim ld.
// lane L: n = L&15, kb = (L>>4)*8; halves 0..7 -> K=kb..kb+7, 8..15 -> K=kb+16..kb+23
__device__ __forceinline__ v16bf load_frag_b(const bf16_t* base, int ld) {
  int lane = threadIdx.x & 31;
  int n    = lane & 15;
  int kb   = (lane >> 4) << 3;
  const bf16_t* p = base + (size_t)n * ld + kb;
  FragU u;
  u.q[0] = *reinterpret_cast<const u32x4*>(p);
  u.q[1] = *reinterpret_cast<const u32x4*>(p + 16);
  return u.v;
}

__device__ __forceinline__ unsigned pack_bf16x2(float lo, float hi) {
  union { bf16_t h[2]; unsigned u; } p;
  p.h[0] = (bf16_t)lo;
  p.h[1] = (bf16_t)hi;
  return p.u;
}

// ---------------------------------------------------------------------------
// Kernel 1: GroupNorm (32 groups over C) + affine, output bf16 xn (B, C, N)
// grid = B*GROUPS blocks, 256 threads. Each block: 16 channels x 512 = 8192 elems
// ---------------------------------------------------------------------------
__global__ __launch_bounds__(256) void groupnorm_bf16_kernel(
    const float* __restrict__ x, const float* __restrict__ gw,
    const float* __restrict__ gb, bf16_t* __restrict__ xn) {
  const int b = blockIdx.x / GROUPS;
  const int g = blockIdx.x % GROUPS;
  const size_t base = ((size_t)b * C_DIM + (size_t)g * 16) * N_DIM;
  const float* xp = x + base;

  float s = 0.f, ss = 0.f;
  for (int i = threadIdx.x; i < 8192; i += 256) {
    float v = xp[i];
    s += v; ss += v * v;
  }
  #pragma unroll
  for (int o = 16; o > 0; o >>= 1) {
    s  += __shfl_xor(s, o, 32);
    ss += __shfl_xor(ss, o, 32);
  }
  __shared__ float rs[8], rss[8];
  int w = threadIdx.x >> 5, lane = threadIdx.x & 31;
  if (lane == 0) { rs[w] = s; rss[w] = ss; }
  __syncthreads();
  s = 0.f; ss = 0.f;
  #pragma unroll
  for (int i = 0; i < 8; i++) { s += rs[i]; ss += rss[i]; }

  const float mean = s * (1.0f / 8192.0f);
  const float var  = ss * (1.0f / 8192.0f) - mean * mean;
  const float rstd = rsqrtf(var + EPS_GN);

  bf16_t* xo = xn + base;
  for (int i = threadIdx.x; i < 8192; i += 256) {
    int c = g * 16 + (i >> 9);  // i / 512
    float v = (xp[i] - mean) * rstd * gw[c] + gb[c];
    xo[i] = (bf16_t)v;
  }
}

// ---------------------------------------------------------------------------
// Kernel 2: f32 -> bf16 convert (weights)
// ---------------------------------------------------------------------------
__global__ __launch_bounds__(256) void cvt_bf16_kernel(
    const float* __restrict__ src, bf16_t* __restrict__ dst, int n) {
  int i = blockIdx.x * 256 + threadIdx.x;
  if (i < n) dst[i] = (bf16_t)src[i];
}

// ---------------------------------------------------------------------------
// Kernel 3: fused QKV GEMM.  q/k/v[b] = xn[b](C x N) @ W^T(N x OUT) + bias
// Block tile 64(M) x 128(N), 8 waves: wave = 32x32, 2x2 WMMA accs per matrix.
// All 8 fragment loads of a K-step are issued before the 12-WMMA chain so the
// scheduler can keep them in flight in one clause.
// Outputs: q,k row-major bf16 (B,C,OUT); v stored TRANSPOSED bf16 (B,OUT,C)
// ---------------------------------------------------------------------------
__global__ __launch_bounds__(256) void qkv_gemm_kernel(
    const bf16_t* __restrict__ xn,
    const bf16_t* __restrict__ wq, const bf16_t* __restrict__ wk,
    const bf16_t* __restrict__ wv,
    const float* __restrict__ qbias, const float* __restrict__ kbias,
    const float* __restrict__ vbias,
    bf16_t* __restrict__ qo, bf16_t* __restrict__ ko, bf16_t* __restrict__ vto) {
  const int b = blockIdx.z;
  const int m_blk = blockIdx.x * 64;    // over C
  const int n_blk = blockIdx.y * 128;   // over OUT
  const int w = threadIdx.x >> 5;
  const int lane = threadIdx.x & 31;
  const int wm = (w & 1) * 32;
  const int n0 = n_blk + (w >> 1) * 32;

  const bf16_t* A = xn + (size_t)b * C_DIM * N_DIM + (size_t)(m_blk + wm) * N_DIM;

  v8f aq[2][2] = {}; v8f ak[2][2] = {}; v8f av[2][2] = {};

  for (int k0 = 0; k0 < N_DIM; k0 += 32) {
    // Issue every load of this K-step first (distinct live ranges).
    v16bf a0  = load_frag_a(A + k0, N_DIM);
    v16bf a1  = load_frag_a(A + (size_t)16 * N_DIM + k0, N_DIM);
    v16bf bq0 = load_frag_b(wq + (size_t)n0 * N_DIM + k0, N_DIM);
    v16bf bq1 = load_frag_b(wq + (size_t)(n0 + 16) * N_DIM + k0, N_DIM);
    v16bf bk0 = load_frag_b(wk + (size_t)n0 * N_DIM + k0, N_DIM);
    v16bf bk1 = load_frag_b(wk + (size_t)(n0 + 16) * N_DIM + k0, N_DIM);
    v16bf bv0 = load_frag_b(wv + (size_t)n0 * N_DIM + k0, N_DIM);
    v16bf bv1 = load_frag_b(wv + (size_t)(n0 + 16) * N_DIM + k0, N_DIM);

    aq[0][0] = wmma_bf16(a0, bq0, aq[0][0]);
    aq[0][1] = wmma_bf16(a0, bq1, aq[0][1]);
    aq[1][0] = wmma_bf16(a1, bq0, aq[1][0]);
    aq[1][1] = wmma_bf16(a1, bq1, aq[1][1]);

    ak[0][0] = wmma_bf16(a0, bk0, ak[0][0]);
    ak[0][1] = wmma_bf16(a0, bk1, ak[0][1]);
    ak[1][0] = wmma_bf16(a1, bk0, ak[1][0]);
    ak[1][1] = wmma_bf16(a1, bk1, ak[1][1]);

    av[0][0] = wmma_bf16(a0, bv0, av[0][0]);
    av[0][1] = wmma_bf16(a0, bv1, av[0][1]);
    av[1][0] = wmma_bf16(a1, bv0, av[1][0]);
    av[1][1] = wmma_bf16(a1, bv1, av[1][1]);
  }

  // Epilogue: D layout: lane L -> col n=L&15 ; VGPR e -> row e + 8*(L>>4)
  const int col  = lane & 15;
  const int rofs = (lane >> 4) * 8;
  bf16_t* qb_ = qo  + (size_t)b * C_DIM * OUT_DIM;
  bf16_t* kb_ = ko  + (size_t)b * C_DIM * OUT_DIM;
  bf16_t* vt_ = vto + (size_t)b * OUT_DIM * C_DIM;
  #pragma unroll
  for (int i = 0; i < 2; i++) {
    #pragma unroll
    for (int j = 0; j < 2; j++) {
      int gm = m_blk + wm + i * 16 + rofs;
      int gn = n0 + j * 16 + col;
      float biq = qbias[gn], bik = kbias[gn], biv = vbias[gn];
      #pragma unroll
      for (int e = 0; e < 8; e++) {
        int row = gm + e;
        qb_[(size_t)row * OUT_DIM + gn] = (bf16_t)(aq[i][j][e] + biq);
        kb_[(size_t)row * OUT_DIM + gn] = (bf16_t)(ak[i][j][e] + bik);
        vt_[(size_t)gn * C_DIM + row]   = (bf16_t)(av[i][j][e] + biv);  // transposed
      }
    }
  }
}

// ---------------------------------------------------------------------------
// Kernel 4: fused attention: scores = q@k^T * scale -> softmax -> @ v
// grid (C/32, B). Block = 256 thr, 8 waves (2 in M x 4 in N). 32 query rows.
// Scores staged in 64KB LDS (f32). Softmax writes normalized probs back
// IN-PLACE as packed bf16 (first 1024B of each 2048B f32 row), so phase-3
// A fragments are two 16B LDS loads (effective bf16 leading dim = 1024).
// ---------------------------------------------------------------------------
__global__ __launch_bounds__(256) void attention_kernel(
    const bf16_t* __restrict__ q, const bf16_t* __restrict__ k,
    const bf16_t* __restrict__ vT, float* __restrict__ out) {
  __shared__ float sS[32 * 512];  // 64 KB

  const int b  = blockIdx.y;
  const int q0 = blockIdx.x * 32;
  const int w  = threadIdx.x >> 5;
  const int lane = threadIdx.x & 31;
  const int wm = (w & 1) * 16;    // row offset within 32-row block
  const int wn = (w >> 1) * 128;  // column strip base
  const int col  = lane & 15;
  const int rofs = (lane >> 4) * 8;
  const float scale = 0.044194173824f;  // 1/sqrt(512)

  const bf16_t* Q  = q  + (size_t)b * C_DIM * OUT_DIM + (size_t)(q0 + wm) * OUT_DIM;
  const bf16_t* K  = k  + (size_t)b * C_DIM * OUT_DIM;
  const bf16_t* VT = vT + (size_t)b * OUT_DIM * C_DIM;

  // ---- Phase 1: scores(16 x 128 per wave) ----
  {
    v8f acc[8] = {};
    for (int k0 = 0; k0 < OUT_DIM; k0 += 32) {
      v16bf a = load_frag_a(Q + k0, OUT_DIM);
      v16bf bf[8];
      #pragma unroll
      for (int t = 0; t < 8; t++)
        bf[t] = load_frag_b(K + (size_t)(wn + t * 16) * OUT_DIM + k0, OUT_DIM);
      #pragma unroll
      for (int t = 0; t < 8; t++)
        acc[t] = wmma_bf16(a, bf[t], acc[t]);
    }
    #pragma unroll
    for (int t = 0; t < 8; t++)
      #pragma unroll
      for (int e = 0; e < 8; e++)
        sS[(wm + rofs + e) * 512 + wn + t * 16 + col] = acc[t][e] * scale;
  }
  __syncthreads();

  // ---- Phase 2: row softmax; each wave owns 4 full rows.
  // Lane handles 8 adjacent column pairs (cols 2p,2p+1 ; p = lane + 32*i):
  // conflict-free float2 reads, then packed-bf16 u32 writes in-place.
  // All reads data-depend on the full-row sum, so they precede the stores.
  for (int r = w; r < 32; r += 8) {
    float* row = sS + r * 512;
    f32x2 vals[8];
    float mx = -3.402823466e38f;
    #pragma unroll
    for (int i = 0; i < 8; i++) {
      vals[i] = *reinterpret_cast<const f32x2*>(&row[2 * lane + 64 * i]);
      mx = fmaxf(mx, fmaxf(vals[i][0], vals[i][1]));
    }
    #pragma unroll
    for (int o = 16; o > 0; o >>= 1) mx = fmaxf(mx, __shfl_xor(mx, o, 32));
    float s = 0.f;
    #pragma unroll
    for (int i = 0; i < 8; i++) {
      vals[i][0] = __expf(vals[i][0] - mx);
      vals[i][1] = __expf(vals[i][1] - mx);
      s += vals[i][0] + vals[i][1];
    }
    #pragma unroll
    for (int o = 16; o > 0; o >>= 1) s += __shfl_xor(s, o, 32);
    float inv = 1.0f / s;
    unsigned* rowp = reinterpret_cast<unsigned*>(row);  // packed bf16 view
    #pragma unroll
    for (int i = 0; i < 8; i++)
      rowp[lane + 32 * i] = pack_bf16x2(vals[i][0] * inv, vals[i][1] * inv);
  }
  __syncthreads();

  // ---- Phase 3: out(16 x 128 per wave) = attn(bf16 in LDS) @ v (transposed) ----
  {
    v8f acc[8] = {};
    // bf16 rows are embedded at the start of each 512-float row => ld = 1024 bf16
    const bf16_t* Abase = reinterpret_cast<const bf16_t*>(sS) + (size_t)wm * 1024;
    for (int k0 = 0; k0 < C_DIM; k0 += 32) {
      v16bf a = load_frag_a(Abase + k0, 1024);
      v16bf bf[8];
      #pragma unroll
      for (int t = 0; t < 8; t++)
        bf[t] = load_frag_b(VT + (size_t)(wn + t * 16) * C_DIM + k0, C_DIM);
      #pragma unroll
      for (int t = 0; t < 8; t++)
        acc[t] = wmma_bf16(a, bf[t], acc[t]);
    }
    float* O = out + (size_t)b * C_DIM * OUT_DIM + (size_t)q0 * OUT_DIM;
    #pragma unroll
    for (int t = 0; t < 8; t++)
      #pragma unroll
      for (int e = 0; e < 8; e++)
        O[(size_t)(wm + rofs + e) * OUT_DIM + wn + t * 16 + col] = acc[t][e];
  }
}

// ---------------------------------------------------------------------------
extern "C" void kernel_launch(void* const* d_in, const int* in_sizes, int n_in,
                              void* d_out, int out_size, void* d_ws, size_t ws_size,
                              hipStream_t stream) {
  const float* x    = (const float*)d_in[0];
  const float* q_w  = (const float*)d_in[1];
  const float* q_b  = (const float*)d_in[2];
  const float* k_w  = (const float*)d_in[3];
  const float* k_b  = (const float*)d_in[4];
  const float* v_w  = (const float*)d_in[5];
  const float* v_b  = (const float*)d_in[6];
  const float* gn_w = (const float*)d_in[7];
  const float* gn_b = (const float*)d_in[8];

  // Workspace layout (~129.5 MB total)
  const size_t XN_BYTES = (size_t)B_SZ * C_DIM * N_DIM * sizeof(bf16_t);  // 32 MB
  const size_t W_BYTES  = (size_t)OUT_DIM * N_DIM * sizeof(bf16_t);       // 0.5 MB
  char* p = (char*)d_ws;
  bf16_t* xn  = (bf16_t*)p; p += XN_BYTES;
  bf16_t* wqb = (bf16_t*)p; p += W_BYTES;
  bf16_t* wkb = (bf16_t*)p; p += W_BYTES;
  bf16_t* wvb = (bf16_t*)p; p += W_BYTES;
  bf16_t* qbuf  = (bf16_t*)p; p += XN_BYTES;
  bf16_t* kbuf  = (bf16_t*)p; p += XN_BYTES;
  bf16_t* vtbuf = (bf16_t*)p; p += XN_BYTES;

  // 1) GroupNorm -> bf16 xn
  groupnorm_bf16_kernel<<<B_SZ * GROUPS, 256, 0, stream>>>(x, gn_w, gn_b, xn);

  // 2) weights f32 -> bf16
  const int wn = OUT_DIM * N_DIM;
  cvt_bf16_kernel<<<(wn + 255) / 256, 256, 0, stream>>>(q_w, wqb, wn);
  cvt_bf16_kernel<<<(wn + 255) / 256, 256, 0, stream>>>(k_w, wkb, wn);
  cvt_bf16_kernel<<<(wn + 255) / 256, 256, 0, stream>>>(v_w, wvb, wn);

  // 3) fused QKV GEMM
  dim3 gq(C_DIM / 64, OUT_DIM / 128, B_SZ);
  qkv_gemm_kernel<<<gq, 256, 0, stream>>>(xn, wqb, wkb, wvb, q_b, k_b, v_b,
                                          qbuf, kbuf, vtbuf);

  // 4) fused attention (scores + softmax + attn@v)
  dim3 ga(C_DIM / 32, B_SZ);
  attention_kernel<<<ga, 256, 0, stream>>>(qbuf, kbuf, vtbuf, (float*)d_out);
}
